// InnocentAttention_29463475651104
// MI455X (gfx1250) — matrix operand order repaired and checked by reading, not compiled, not run on hardware
//
#include <hip/hip_runtime.h>
#include <hip/hip_bf16.h>

typedef _Float16 v16h __attribute__((ext_vector_type(16)));
typedef _Float16 v8h  __attribute__((ext_vector_type(8)));
typedef float    v8f  __attribute__((ext_vector_type(8)));
typedef float    v4f  __attribute__((ext_vector_type(4)));
typedef int      v4i  __attribute__((ext_vector_type(4)));

typedef __attribute__((address_space(1))) v4i global_v4i;
typedef __attribute__((address_space(3))) v4i shared_v4i;

#define NEGV (-1.0e9f)

#if defined(__has_builtin)
#  if __has_builtin(__builtin_amdgcn_global_load_async_to_lds_b128)
#    define HAVE_ASYNC_BUILTIN 1
#  endif
#  if __has_builtin(__builtin_amdgcn_s_wait_asynccnt)
#    define HAVE_ASYNCCNT_BUILTIN 1
#  endif
#endif

// One b128 (16B) async copy global -> LDS, tracked by ASYNCcnt.
__device__ __forceinline__ void async_cp_b128(const float* g, float* l)
{
#ifdef HAVE_ASYNC_BUILTIN
    __builtin_amdgcn_global_load_async_to_lds_b128(
        (global_v4i*)g, (shared_v4i*)l, 0, 0);
#else
    unsigned lds_off =
        (unsigned)(size_t)(__attribute__((address_space(3))) void*)l;
    asm volatile("global_load_async_to_lds_b128 %0, %1, off"
                 :: "v"(lds_off), "v"((unsigned long long)(size_t)g)
                 : "memory");
#endif
}

__device__ __forceinline__ void wait_async0()
{
#ifdef HAVE_ASYNCCNT_BUILTIN
    __builtin_amdgcn_s_wait_asynccnt(0);
#else
    asm volatile("s_wait_asynccnt 0" ::: "memory");
#endif
}

// ---- Cross-lane xor-exchange within 16-lane halves, pure VALU (DPP) ----
#define DPP8_SEL(s0,s1,s2,s3,s4,s5,s6,s7) \
    ((s0)|((s1)<<3)|((s2)<<6)|((s3)<<9)|((s4)<<12)|((s5)<<15)|((s6)<<18)|((s7)<<21))

template <int SEL>
__device__ __forceinline__ float dpp8f(float x)
{
    return __int_as_float(
        __builtin_amdgcn_mov_dpp8(__float_as_int(x), SEL));
}

__device__ __forceinline__ float dpp_xor8(float x)
{
    // DPP16 ROW_XMASK:8 (0x160|8): lane <- lane ^ 8 within each row of 16
    return __int_as_float(__builtin_amdgcn_update_dpp(
        __float_as_int(x), __float_as_int(x), 0x168, 0xf, 0xf, true));
}

// max across the 16 lanes of each wave half (columns of one C-layout row)
__device__ __forceinline__ float rowmax16(float x)
{
    x = fmaxf(x, dpp8f<DPP8_SEL(1,0,3,2,5,4,7,6)>(x));  // xor 1
    x = fmaxf(x, dpp8f<DPP8_SEL(2,3,0,1,6,7,4,5)>(x));  // xor 2
    x = fmaxf(x, dpp8f<DPP8_SEL(4,5,6,7,0,1,2,3)>(x));  // xor 4
    x = fmaxf(x, dpp_xor8(x));                           // xor 8
    return x;
}

// B=4, H=16, S=2048, D=64. One workgroup = 8 waves, each wave owns 16 query
// rows (128 q rows / block). Flash-attention over 32-key blocks using
// v_wmma_f32_16x16x32_f16 for QK^T, PV, and the softmax row-sum (P*ones),
// with ASYNCcnt-pipelined global->LDS staging of the K/V tiles.
__global__ __launch_bounds__(256)
void fa_fwd_kernel(const float* __restrict__ Q,
                   const float* __restrict__ K,
                   const float* __restrict__ V,
                   const int*   __restrict__ EL,
                   float*       __restrict__ O)
{
    constexpr int S = 2048, D = 64, H = 16;

    const int tid  = threadIdx.x;
    const int lane = tid & 31;
    const int wave = tid >> 5;
    const int ln   = lane & 15;   // column index within 16-lane half
    const int lh   = lane >> 4;   // which half of the wave

    const int nqb = S / 128;
    const int bh  = blockIdx.x / nqb;
    const int qb  = blockIdx.x % nqb;
    const int b   = bh / H;
    const int evl = EL[b];

    const float* Qb = Q + (size_t)bh * S * D;
    const float* Kb = K + (size_t)bh * S * D;
    const float* Vb = V + (size_t)bh * S * D;
    float*       Ob = O + (size_t)bh * S * D;

    // Ping-pong raw fp32 tiles filled by async DMA, plus f16 operand tiles.
    __shared__ __align__(16) float    rawK[2][32 * 64];  // 2 x 8KB
    __shared__ __align__(16) float    rawV[2][32 * 64];  // 2 x 8KB
    __shared__ __align__(32) _Float16 sK [32 * 64];      // K tile f16, [key][d]
    __shared__ __align__(32) _Float16 sVt[64 * 32];      // V^T tile f16, [d][key]
    __shared__ __align__(32) _Float16 sP [8][16 * 32];   // per-wave P scratch

    const int q0   = qb * 128 + wave * 16;  // first query row of this wave
    const int qrow = q0 + ln;

    // ---- Load Q tile into WMMA A layout (scale 1/sqrt(D)=0.125 folded in) ----
    // 16-bit A 16x32: lanes 0-15 hold K = {0..7, 16..23}, lanes 16-31 {8..15, 24..31}.
    v16h Aq[2];
    #pragma unroll
    for (int c = 0; c < 2; ++c) {
        const float* src = Qb + (size_t)qrow * D + c * 32 + lh * 8;
        v4f f0 = *(const v4f*)(src);
        v4f f1 = *(const v4f*)(src + 4);
        v4f f2 = *(const v4f*)(src + 16);
        v4f f3 = *(const v4f*)(src + 20);
        #pragma unroll
        for (int i = 0; i < 4; ++i) {
            Aq[c][i]      = (_Float16)(f0[i] * 0.125f);
            Aq[c][4 + i]  = (_Float16)(f1[i] * 0.125f);
            Aq[c][8 + i]  = (_Float16)(f2[i] * 0.125f);
            Aq[c][12 + i] = (_Float16)(f3[i] * 0.125f);
        }
    }

    // All-ones B matrix for the WMMA row-sum trick.
    v16h Bones;
    #pragma unroll
    for (int i = 0; i < 16; ++i) Bones[i] = (_Float16)1.0f;

    // ---- Online-softmax state (C layout: VGPR r = row r (+8 for upper half)) ----
    float mrow[8], lrow[8];
    v8f Oacc[4];
    #pragma unroll
    for (int r = 0; r < 8; ++r) { mrow[r] = -__builtin_inff(); lrow[r] = 0.0f; }
    #pragma unroll
    for (int j = 0; j < 4; ++j)
        #pragma unroll
        for (int e = 0; e < 8; ++e) Oacc[j][e] = 0.0f;

    // K/V 32-key tiles are contiguous 8KB blocks (row stride == D == 64).
    // Each thread DMAs 32B of K and 32B of V per tile (4 async b128 ops).
    const int toff = tid * 8;
    auto stage_async = [&](int buf, int kb) {
        const float* gk = Kb + (size_t)kb * D + toff;
        const float* gv = Vb + (size_t)kb * D + toff;
        async_cp_b128(gk,     &rawK[buf][toff]);
        async_cp_b128(gk + 4, &rawK[buf][toff + 4]);
        async_cp_b128(gv,     &rawV[buf][toff]);
        async_cp_b128(gv + 4, &rawV[buf][toff + 4]);
    };

    // Prologue: kick off DMA for tile 0.
    stage_async(0, 0);

    for (int it = 0; it < S / 32; ++it) {
        const int kb  = it * 32;
        const int buf = it & 1;

        wait_async0();       // this wave's DMA for tile `it` has landed in LDS
        __syncthreads();     // all waves' DMA done; previous compute done

        // Kick off DMA for tile it+1 (overlaps convert + compute below).
        if (kb + 32 < S) stage_async(buf ^ 1, kb + 32);

        // Cache-prefetch two tiles ahead (global_prefetch_b8).
        if (kb + 64 < S) {
            __builtin_prefetch(Kb + (size_t)(kb + 64) * D + toff, 0, 1);
            __builtin_prefetch(Vb + (size_t)(kb + 64) * D + toff, 0, 1);
        }

        // ---- Convert raw fp32 tile `it` -> f16 operand tiles ----
        {   // K: [key][d], coalesced b128 reads, b128 store
            const int key = tid >> 3;
            const int dd  = (tid & 7) << 3;
            v4f a0 = *(const v4f*)&rawK[buf][key * 64 + dd];
            v4f a1 = *(const v4f*)&rawK[buf][key * 64 + dd + 4];
            v8h h;
            #pragma unroll
            for (int i = 0; i < 4; ++i) {
                h[i]     = (_Float16)a0[i];
                h[4 + i] = (_Float16)a1[i];
            }
            *(v8h*)&sK[key * 64 + dd] = h;
        }
        {   // V: transpose to [d][key]; conflict-free b32 reads, b128 store
            const int dc = tid & 63;
            const int k0 = (tid >> 6) << 3;
            v8h h;
            #pragma unroll
            for (int i = 0; i < 8; ++i)
                h[i] = (_Float16)rawV[buf][(k0 + i) * 64 + dc];
            *(v8h*)&sVt[dc * 32 + k0] = h;
        }
        __syncthreads();

        // ---- S = Q K^T : two 16-key N-tiles, contraction d=64 in two chunks ----
        // 16-bit B 32x16: lanes 0-15 hold col N=lane, K={0..7,16..23}; lanes 16-31 K={8..15,24..31}.
        v8f Stile[2];
        #pragma unroll
        for (int j = 0; j < 2; ++j) {
            v8f acc = {};
            #pragma unroll
            for (int c = 0; c < 2; ++c) {
                v8h bkl = *(const v8h*)&sK[(j * 16 + ln) * 64 + c * 32 + lh * 8];
                v8h bkh = *(const v8h*)&sK[(j * 16 + ln) * 64 + c * 32 + 16 + lh * 8];
                v16h Bk = __builtin_shufflevector(bkl, bkh,
                              0, 1, 2, 3, 4, 5, 6, 7, 8, 9, 10, 11, 12, 13, 14, 15);
                acc = __builtin_amdgcn_wmma_f32_16x16x32_f16(
                          false, Aq[c], false, Bk, (short)0, acc, false, false);
            }
            Stile[j] = acc;
        }

        const bool kv0 = (kb + ln)      < evl;
        const bool kv1 = (kb + 16 + ln) < evl;

        // ---- Mask + online softmax in C layout (max via DPP, no LDS) ----
        float alphaA[8];
        #pragma unroll
        for (int r = 0; r < 8; ++r) {
            const int  qr = q0 + lh * 8 + r;
            const bool qv = qr < evl;
            float s0 = (qv && kv0) ? Stile[0][r] : NEGV;
            float s1 = (qv && kv1) ? Stile[1][r] : NEGV;

            const float mx    = rowmax16(fmaxf(s0, s1));
            const float mprev = mrow[r];
            const float mn    = fmaxf(mprev, mx);
            alphaA[r] = __expf(mprev - mn);
            mrow[r]   = mn;

            const float p0 = __expf(s0 - mn);
            const float p1 = __expf(s1 - mn);

            #pragma unroll
            for (int j = 0; j < 4; ++j) Oacc[j][r] *= alphaA[r];

            // spill P (C layout) to per-wave LDS tile, f16
            const int prow = lh * 8 + r;
            sP[wave][prow * 32 + ln]      = (_Float16)p0;
            sP[wave][prow * 32 + 16 + ln] = (_Float16)p1;
        }

        // ---- Re-read P in A layout (same-wave DS ops are in order) ----
        v8h pl = *(const v8h*)&sP[wave][ln * 32 + lh * 8];
        v8h ph = *(const v8h*)&sP[wave][ln * 32 + 16 + lh * 8];
        v16h Pa = __builtin_shufflevector(pl, ph,
                      0, 1, 2, 3, 4, 5, 6, 7, 8, 9, 10, 11, 12, 13, 14, 15);

        // ---- Row sums via WMMA: Lsum = P * ones (replicated across the row) ----
        v8f Lz = {};
        v8f Lsum = __builtin_amdgcn_wmma_f32_16x16x32_f16(
                       false, Pa, false, Bones, (short)0, Lz, false, false);
        #pragma unroll
        for (int r = 0; r < 8; ++r)
            lrow[r] = lrow[r] * alphaA[r] + Lsum[r];

        // ---- O += P V : contraction = 32 keys, four 16-wide d tiles ----
        #pragma unroll
        for (int j = 0; j < 4; ++j) {
            v8h bvl = *(const v8h*)&sVt[(j * 16 + ln) * 32 + lh * 8];
            v8h bvh = *(const v8h*)&sVt[(j * 16 + ln) * 32 + 16 + lh * 8];
            v16h Bv = __builtin_shufflevector(bvl, bvh,
                          0, 1, 2, 3, 4, 5, 6, 7, 8, 9, 10, 11, 12, 13, 14, 15);
            Oacc[j] = __builtin_amdgcn_wmma_f32_16x16x32_f16(
                          false, Pa, false, Bv, (short)0, Oacc[j], false, false);
        }
    }

    // ---- Epilogue: normalize by row sum, store fp32 ----
    #pragma unroll
    for (int r = 0; r < 8; ++r) {
        const int   qr  = q0 + lh * 8 + r;
        const float inv = 1.0f / lrow[r];
        float* dst = Ob + (size_t)qr * D;
        #pragma unroll
        for (int j = 0; j < 4; ++j)
            dst[j * 16 + ln] = Oacc[j][r] * inv;
    }
}

extern "C" void kernel_launch(void* const* d_in, const int* in_sizes, int n_in,
                              void* d_out, int out_size, void* d_ws, size_t ws_size,
                              hipStream_t stream)
{
    (void)in_sizes; (void)n_in; (void)out_size; (void)d_ws; (void)ws_size;
    const float* q   = (const float*)d_in[0];
    const float* k   = (const float*)d_in[1];
    const float* v   = (const float*)d_in[2];
    const int*   evl = (const int*)d_in[3];
    float* out = (float*)d_out;

    // grid: B*H*(S/128) = 4*16*16 = 1024 workgroups of 256 threads (8 waves)
    dim3 grid(4 * 16 * (2048 / 128));
    fa_fwd_kernel<<<grid, 256, 0, stream>>>(q, k, v, evl, out);
}
